// RNNModel_19937238188803
// MI455X (gfx1250) — matrix operand order repaired and checked by reading, not compiled, not run on hardware
//
#include <hip/hip_runtime.h>
#include <hip/hip_bf16.h>
#include <math.h>

#define NTOKEN 50000
#define NINP   400
#define NINP_P 416
#define NHID   1150
#define NHID_P 1152
#define TT     70
#define BB     80
#define NS     16
#define NPOS   (TT*BB)        // 5600
#define NNEG   (NS*NPOS)      // 89600
#define CTILES (NHID_P/16)    // 72

typedef __bf16 bf16_t;
typedef __attribute__((ext_vector_type(16))) __bf16 v16bf;
typedef __attribute__((ext_vector_type(8)))  float  v8f;

// ---- WMMA fragment loaders (bf16 16x16x32, wave32) ----------------------
// A (16xK tile): lanes 0-15 hold M=lane, K = {kk+0..7, kk+16..23};
//                lanes 16-31 hold M=lane-16, K = {kk+8..15, kk+24..31}.
__device__ inline v16bf load_a_bf16(const bf16_t* A, int lda, int rowBase,
                                    int kk, int lane) {
  int r  = rowBase + (lane & 15);
  int kh = (lane >> 4) << 3;
  const bf16_t* p = A + (size_t)r * lda + kk + kh;
  union { v16bf v; float4 q[2]; } u;
  u.q[0] = *(const float4*)(p);
  u.q[1] = *(const float4*)(p + 16);
  return u.v;
}

// B (Kx16 tile) from row-major W[n][k] (B[k][n] = W[n][k]):
// lanes 0-15: N=lane, K={kk+0..7, kk+16..23}; lanes 16-31: N=lane-16, K={kk+8..15, kk+24..31}.
__device__ inline v16bf load_b_bf16(const bf16_t* W, int ldw, int colBase,
                                    int kk, int lane) {
  int n  = colBase + (lane & 15);
  int ko = kk + ((lane >> 4) << 3);
  const bf16_t* p = W + (size_t)n * ldw + ko;
  union { v16bf v; float4 q[2]; } u;
  u.q[0] = *(const float4*)(p);
  u.q[1] = *(const float4*)(p + 16);
  return u.v;
}

// Pack 16 f32 into a bf16 A-fragment (register-only converts).
__device__ inline v16bf pack16(const float4 q0, const float4 q1,
                               const float4 q2, const float4 q3) {
  union { v16bf v; bf16_t e[16]; } u;
  u.e[0]  = (bf16_t)q0.x; u.e[1]  = (bf16_t)q0.y;
  u.e[2]  = (bf16_t)q0.z; u.e[3]  = (bf16_t)q0.w;
  u.e[4]  = (bf16_t)q1.x; u.e[5]  = (bf16_t)q1.y;
  u.e[6]  = (bf16_t)q1.z; u.e[7]  = (bf16_t)q1.w;
  u.e[8]  = (bf16_t)q2.x; u.e[9]  = (bf16_t)q2.y;
  u.e[10] = (bf16_t)q2.z; u.e[11] = (bf16_t)q2.w;
  u.e[12] = (bf16_t)q3.x; u.e[13] = (bf16_t)q3.y;
  u.e[14] = (bf16_t)q3.z; u.e[15] = (bf16_t)q3.w;
  return u.v;
}

// Full in-bounds chunk (kk <= 352): unconditional float4 loads.
__device__ inline v16bf load_a_gather_full(const float* row, int kk, int lane) {
  int kh = (lane >> 4) << 3;
  const float* p = row + kk + kh;
  return pack16(*(const float4*)(p),      *(const float4*)(p + 4),
                *(const float4*)(p + 16), *(const float4*)(p + 20));
}

// Tail chunk kk=384: k = 384+kh..391+kh valid (<400), k = 400+kh.. all pad -> 0.
__device__ inline v16bf load_a_gather_tail(const float* row, int lane) {
  int kh = (lane >> 4) << 3;
  const float* p = row + 384 + kh;
  const float4 z = {0.f, 0.f, 0.f, 0.f};
  return pack16(*(const float4*)(p), *(const float4*)(p + 4), z, z);
}

#define WMMA_BF16(c, a, b) \
  c = __builtin_amdgcn_wmma_f32_16x16x32_bf16(false, (a), false, (b), (short)0, (c), false, false)

// Multi-column-tile inner loops -------------------------------------------
template <int NC>
__device__ inline void loop_gather(const float* arow, const bf16_t* W,
                                   int colBase, int lane, v8f c[NC]) {
  for (int kk = 0; kk < 384; kk += 32) {
    v16bf a = load_a_gather_full(arow, kk, lane);
#pragma unroll
    for (int cc = 0; cc < NC; ++cc) {
      v16bf b = load_b_bf16(W, NINP_P, colBase + cc * 16, kk, lane);
      WMMA_BF16(c[cc], a, b);
    }
  }
  v16bf a = load_a_gather_tail(arow, lane);
#pragma unroll
  for (int cc = 0; cc < NC; ++cc) {
    v16bf b = load_b_bf16(W, NINP_P, colBase + cc * 16, 384, lane);
    WMMA_BF16(c[cc], a, b);
  }
}

template <int NC>
__device__ inline void loop_bf16(const bf16_t* A, int lda, int rowBase,
                                 const bf16_t* W, int ldw, int colBase,
                                 int K, int lane, v8f c[NC]) {
  for (int kk = 0; kk < K; kk += 32) {
    v16bf a = load_a_bf16(A, lda, rowBase, kk, lane);
#pragma unroll
    for (int cc = 0; cc < NC; ++cc) {
      v16bf b = load_b_bf16(W, ldw, colBase + cc * 16, kk, lane);
      WMMA_BF16(c[cc], a, b);
    }
  }
}

// ---- conversion / init kernels ------------------------------------------
__global__ void k_conv_pad(const float* __restrict__ src, bf16_t* __restrict__ dst,
                           int R, int C, int Cp, int total) {
  int idx = blockIdx.x * blockDim.x + threadIdx.x;
  if (idx >= total) return;
  int r = idx / Cp, c = idx % Cp;
  float v = (r < R && c < C) ? src[(size_t)r * C + c] : 0.f;
  dst[idx] = (bf16_t)v;
}

__global__ void k_init_h0(const float* __restrict__ hidden,
                          float* __restrict__ st0, bf16_t* __restrict__ pb0) {
  int idx = blockIdx.x * blockDim.x + threadIdx.x;
  if (idx >= BB * NHID_P) return;
  int b = idx / NHID_P, h = idx % NHID_P;
  float v = (h < NHID) ? hidden[(size_t)b * NHID + h] : 0.f;
  st0[idx] = v;
  pb0[idx] = (bf16_t)v;
}

// ---- xW = emb_W[data] @ W_ih^T + b_ih  (gather-A WMMA GEMM, 16x64 tiles) -
__global__ void k_gemm_embed(const int* __restrict__ tokens,
                             const float* __restrict__ embW,
                             const bf16_t* __restrict__ Wb,
                             const float* __restrict__ bias,
                             float* __restrict__ out) {
  constexpr int NC = 4;
  int rt = blockIdx.x, lane = threadIdx.x;
  int rowBase = rt * 16, colBase = blockIdx.y * 16 * NC;
  int tok = tokens[rowBase + (lane & 15)];
  const float* arow = embW + (size_t)tok * NINP;
  v8f c[NC] = {};
  loop_gather<NC>(arow, Wb, colBase, lane, c);
  int rbase = rowBase + ((lane >> 4) << 3);
#pragma unroll
  for (int cc = 0; cc < NC; ++cc) {
    int col = colBase + cc * 16 + (lane & 15);
    float bv = (col < NHID) ? bias[col] : 0.f;
#pragma unroll
    for (int i = 0; i < 8; ++i)
      out[(size_t)(rbase + i) * NHID_P + col] = c[cc][i] + bv;
  }
}

// ---- recurrent step: h' = tanh(h@W_hh^T + xW_t + b_hh); writes f32 + bf16
__global__ void k_scan_step(const bf16_t* __restrict__ hbf,
                            const bf16_t* __restrict__ Whh_b,
                            const float* __restrict__ xW_t,
                            const float* __restrict__ b_hh,
                            float* __restrict__ st_out,
                            bf16_t* __restrict__ hbf_out) {
  constexpr int NC = 2;
  int rt = blockIdx.x, lane = threadIdx.x;
  int rowBase = rt * 16, colBase = blockIdx.y * 16 * NC;
  v8f c[NC] = {};
  loop_bf16<NC>(hbf, NHID_P, rowBase, Whh_b, NHID_P, colBase, NHID_P, lane, c);
  int rbase = rowBase + ((lane >> 4) << 3);
#pragma unroll
  for (int cc = 0; cc < NC; ++cc) {
    int col = colBase + cc * 16 + (lane & 15);
    float bv = (col < NHID) ? b_hh[col] : 0.f;
#pragma unroll
    for (int i = 0; i < 8; ++i) {
      size_t idx = (size_t)(rbase + i) * NHID_P + col;
      float v = tanhf(c[cc][i] + xW_t[idx] + bv);
      st_out[idx]  = v;
      hbf_out[idx] = (bf16_t)v;
    }
  }
}

// ---- hU = prev @ W_hh^T + b_hh  (bf16 WMMA GEMM, 16x64 tiles) -----------
__global__ void k_gemm_hU(const bf16_t* __restrict__ Abf,
                          const bf16_t* __restrict__ Whh_b,
                          const float* __restrict__ b_hh,
                          float* __restrict__ out) {
  constexpr int NC = 4;
  int rt = blockIdx.x, lane = threadIdx.x;
  int rowBase = rt * 16, colBase = blockIdx.y * 16 * NC;
  v8f c[NC] = {};
  loop_bf16<NC>(Abf, NHID_P, rowBase, Whh_b, NHID_P, colBase, NHID_P, lane, c);
  int rbase = rowBase + ((lane >> 4) << 3);
#pragma unroll
  for (int cc = 0; cc < NC; ++cc) {
    int col = colBase + cc * 16 + (lane & 15);
    float bv = (col < NHID) ? b_hh[col] : 0.f;
#pragma unroll
    for (int i = 0; i < 8; ++i)
      out[(size_t)(rbase + i) * NHID_P + col] = c[cc][i] + bv;
  }
}

// ---- fused negative-sample GEMM + tanh + squared-distance partials ------
__global__ void k_neg(const int* __restrict__ samples,
                      const float* __restrict__ embW,
                      const bf16_t* __restrict__ Wih_b,
                      const float* __restrict__ b_ih,
                      const float* __restrict__ hU,
                      const float* __restrict__ prev,   // states[0..T-1] = [5600][1152]
                      float* __restrict__ dpart) {      // [CTILES][NNEG]
  constexpr int NC = 4;
  int rt = blockIdx.x, lane = threadIdx.x;
  int rowBase = rt * 16, colBase = blockIdx.y * 16 * NC;
  int tok = samples[rowBase + (lane & 15)];
  const float* arow = embW + (size_t)tok * NINP;
  v8f c[NC] = {};
  loop_gather<NC>(arow, Wih_b, colBase, lane, c);

  int rbase = rowBase + ((lane >> 4) << 3);
#pragma unroll
  for (int cc = 0; cc < NC; ++cc) {
    int col = colBase + cc * 16 + (lane & 15);
    float bv = (col < NHID) ? b_ih[col] : 0.f;
    float dsum[8];
#pragma unroll
    for (int i = 0; i < 8; ++i) {
      int row = rbase + i;              // [0, 89600)
      int n   = row % NPOS;
      float o = tanhf(c[cc][i] + bv + hU[(size_t)n * NHID_P + col]);
      float p = prev[(size_t)n * NHID_P + col];
      float d = p - o;
      dsum[i] = d * d;
    }
    // reduce over 16 columns: lanes 0-15 hold rows 0-7, lanes 16-31 rows 8-15;
    // xor masks < 16 keep halves independent.
#pragma unroll
    for (int m = 1; m < 16; m <<= 1) {
#pragma unroll
      for (int i = 0; i < 8; ++i) dsum[i] += __shfl_xor(dsum[i], m, 32);
    }
    if ((lane & 15) == 0) {
      int ct = colBase / 16 + cc;
#pragma unroll
      for (int i = 0; i < 8; ++i)
        dpart[(size_t)ct * NNEG + (rbase + i)] = dsum[i];
    }
  }
}

__global__ void k_neg_reduce(const float* __restrict__ dpart,
                             float* __restrict__ dneg) {
  int row = blockIdx.x * blockDim.x + threadIdx.x;
  if (row >= NNEG) return;
  float s = 0.f;
  for (int c = 0; c < CTILES; ++c) s += dpart[(size_t)c * NNEG + row];
  dneg[row] = s;
}

// ---- positive distances: x0[n] = -(d_pos + dec_bias[data[n]]) -----------
__global__ void k_dpos(const float* __restrict__ states,
                       const int* __restrict__ data,
                       const float* __restrict__ decb,
                       float* __restrict__ x0) {
  int n = blockIdx.x, lane = threadIdx.x;
  const float* a = states + (size_t)n * NHID_P;
  const float* b = states + (size_t)(n + BB) * NHID_P;   // next timestep, same batch
  float s = 0.f;
  for (int h = lane; h < NHID; h += 32) { float d = b[h] - a[h]; s += d * d; }
#pragma unroll
  for (int m = 1; m < 32; m <<= 1) s += __shfl_xor(s, m, 32);
  if (lane == 0) x0[n] = -(s + decb[data[n]]);
}

// ---- per-position log-softmax NLL ---------------------------------------
__global__ void k_nll(const float* __restrict__ x0,
                      const float* __restrict__ dneg,
                      const int* __restrict__ samples,
                      const float* __restrict__ decb,
                      const float* __restrict__ binary,
                      float* __restrict__ nll) {
  int n = blockIdx.x * blockDim.x + threadIdx.x;
  if (n >= NPOS) return;
  float x[1 + NS];
  x[0] = x0[n];
  float mx = x[0];
#pragma unroll
  for (int s = 0; s < NS; ++s) {
    int tok = samples[s * NPOS + n];
    x[1 + s] = -(dneg[s * NPOS + n] + decb[tok]);
    mx = fmaxf(mx, x[1 + s]);
  }
  float sum = 0.f;
#pragma unroll
  for (int i = 0; i < 1 + NS; ++i) sum += expf(x[i] - mx);
  float lse = mx + logf(sum);
  nll[n] = (lse - x[0]) * binary[n];
}

__global__ void k_reduce(const float* __restrict__ nll, float* __restrict__ out) {
  __shared__ float sm[256];
  float s = 0.f;
  for (int i = threadIdx.x; i < NPOS; i += 256) s += nll[i];
  sm[threadIdx.x] = s;
  __syncthreads();
  for (int k = 128; k > 0; k >>= 1) {
    if (threadIdx.x < k) sm[threadIdx.x] += sm[threadIdx.x + k];
    __syncthreads();
  }
  if (threadIdx.x == 0) out[0] = sm[0] / (float)NPOS;
}

// ---- host orchestration --------------------------------------------------
extern "C" void kernel_launch(void* const* d_in, const int* in_sizes, int n_in,
                              void* d_out, int out_size, void* d_ws, size_t ws_size,
                              hipStream_t stream) {
  (void)in_sizes; (void)n_in; (void)out_size; (void)ws_size;
  const int*   data    = (const int*)  d_in[0];
  const float* binary  = (const float*)d_in[1];
  const float* hidden  = (const float*)d_in[2];
  const int*   samples = (const int*)  d_in[3];
  const float* embW    = (const float*)d_in[4];
  const float* W_ih    = (const float*)d_in[5];
  const float* b_ih    = (const float*)d_in[6];
  const float* W_hh    = (const float*)d_in[7];
  const float* b_hh    = (const float*)d_in[8];
  const float* decb    = (const float*)d_in[9];

  char* ws = (char*)d_ws;
  size_t off = 0;
  auto take = [&](size_t bytes) -> char* {
    char* p = ws + off;
    off += (bytes + 255) & ~(size_t)255;
    return p;
  };
  bf16_t* Wih_b  = (bf16_t*)take((size_t)NHID_P * NINP_P * 2);
  bf16_t* Whh_b  = (bf16_t*)take((size_t)NHID_P * NHID_P * 2);
  float*  states = (float*) take((size_t)(TT + 1) * BB * NHID_P * 4);
  bf16_t* prevbf = (bf16_t*)take((size_t)(TT + 1) * BB * NHID_P * 2);
  float*  xW     = (float*) take((size_t)NPOS * NHID_P * 4);
  float*  hU     = (float*) take((size_t)NPOS * NHID_P * 4);
  float*  x0     = (float*) take((size_t)NPOS * 4);
  float*  dpart  = (float*) take((size_t)CTILES * NNEG * 4);
  float*  dneg   = (float*) take((size_t)NNEG * 4);
  float*  nllb   = (float*) take((size_t)NPOS * 4);

  // weight conversion + padding (pad rows/cols = 0)
  {
    int tot = NHID_P * NINP_P;
    k_conv_pad<<<(tot + 255) / 256, 256, 0, stream>>>(W_ih, Wih_b, NHID, NINP, NINP_P, tot);
    tot = NHID_P * NHID_P;
    k_conv_pad<<<(tot + 255) / 256, 256, 0, stream>>>(W_hh, Whh_b, NHID, NHID, NHID_P, tot);
  }
  // h0 into padded f32 + bf16
  {
    int tot = BB * NHID_P;
    k_init_h0<<<(tot + 255) / 256, 256, 0, stream>>>(hidden, states, prevbf);
  }
  // xW = emb @ W_ih^T + b_ih   (16x64 tiles per wave)
  k_gemm_embed<<<dim3(NPOS / 16, CTILES / 4), 32, 0, stream>>>(data, embW, Wih_b, b_ih, xW);

  // recurrent scan (70 sequential steps, 16x32 tiles per wave)
  for (int t = 0; t < TT; ++t) {
    k_scan_step<<<dim3(BB / 16, CTILES / 2), 32, 0, stream>>>(
        prevbf + (size_t)t * BB * NHID_P, Whh_b,
        xW + (size_t)t * BB * NHID_P, b_hh,
        states + (size_t)(t + 1) * BB * NHID_P,
        prevbf + (size_t)(t + 1) * BB * NHID_P);
  }

  // hU = prev @ W_hh^T + b_hh
  k_gemm_hU<<<dim3(NPOS / 16, CTILES / 4), 32, 0, stream>>>(prevbf, Whh_b, b_hh, hU);

  // positive distances
  k_dpos<<<NPOS, 32, 0, stream>>>(states, data, decb, x0);

  // fused negative-sample GEMM + distance partials, then reduce
  k_neg<<<dim3(NNEG / 16, CTILES / 4), 32, 0, stream>>>(samples, embW, Wih_b, b_ih, hU, states, dpart);
  k_neg_reduce<<<(NNEG + 255) / 256, 256, 0, stream>>>(dpart, dneg);

  // per-position NLL + mean
  k_nll<<<(NPOS + 255) / 256, 256, 0, stream>>>(x0, dneg, samples, decb, binary, nllb);
  k_reduce<<<1, 256, 0, stream>>>(nllb, (float*)d_out);
}
